// ScalableSageEncoder_78735340470387
// MI455X (gfx1250) — matrix operand order repaired and checked by reading, not compiled, not run on hardware
//
#include <hip/hip_runtime.h>
#include <hip/hip_bf16.h>

#define BATCH 8192
#define FAN   10
#define DIM   256

typedef _Float16 h8   __attribute__((ext_vector_type(8)));
typedef _Float16 v16h __attribute__((ext_vector_type(16)));
typedef float    v8f  __attribute__((ext_vector_type(8)));

// ---------------------------------------------------------------------------
// Weight prep: fp32 row-major W[k][n] (256x256) -> f16, pre-swizzled into the
// WMMA B-operand fragment layout so each lane's fragment is 32 contiguous
// bytes. Half index layout: (((tn*8 + ks)*32 + lane)*16 + h), where
//   n = tn*16 + (lane&15),  K = ks*32 + (lane>>4)*8 + (h&7) + (h>>3)*16
// (CDNA5 ISA: B 32x16 f16 = 8 VGPRs, lane n holds column n, VGPR j packs
//  K=2j,2j+1 (j<4) or 2j+8,2j+9 (j>=4); lanes 16-31 carry K+8.)
// ---------------------------------------------------------------------------
__global__ __launch_bounds__(256) void prep_weights(
    const float* __restrict__ w0, const float* __restrict__ w1,
    const float* __restrict__ w2, const float* __restrict__ w3,
    _Float16* __restrict__ p0, _Float16* __restrict__ p1,
    _Float16* __restrict__ p2, _Float16* __restrict__ p3) {
  int idx = blockIdx.x * blockDim.x + threadIdx.x;   // 0 .. 4*65536-1
  int w = idx >> 16;
  int e = idx & 0xFFFF;
  int h  = e & 15;
  int L  = (e >> 4) & 31;
  int ks = (e >> 9) & 7;
  int tn = e >> 12;
  int K = ks * 32 + (L >> 4) * 8 + (h & 7) + (h >> 3) * 16;
  int n = tn * 16 + (L & 15);
  const float* src = (w == 0) ? w0 : (w == 1) ? w1 : (w == 2) ? w2 : w3;
  _Float16*    dst = (w == 0) ? p0 : (w == 1) ? p1 : (w == 2) ? p2 : p3;
  dst[e] = (_Float16)src[K * DIM + n];
}

// ---------------------------------------------------------------------------
// Gather + fan-in mean. One block per batch row, one thread per feature.
// Streams feature_table/store0 rows once (fully coalesced 1KB row reads) and
// emits f16 activations for the WMMA GEMMs. This is the HBM-bound phase
// (~185 MB -> ~8 us at 23.3 TB/s).
// ---------------------------------------------------------------------------
__global__ __launch_bounds__(DIM) void gather_mean(
    const int* __restrict__ nodes, const int* __restrict__ neighbor,
    const float* __restrict__ ftab, const float* __restrict__ store0,
    _Float16* __restrict__ Xself, _Float16* __restrict__ Xn0,
    _Float16* __restrict__ Xn1) {
  int b = blockIdx.x;
  int d = threadIdx.x;
  int node = nodes[b];
  float x = ftab[(size_t)node * DIM + d];
  float s0 = 0.f, s1 = 0.f;
#pragma unroll
  for (int f = 0; f < FAN; ++f) {
    int idx = neighbor[b * FAN + f];
    size_t off = (size_t)idx * DIM + d;
    s0 += ftab[off];
    s1 += store0[off];
  }
  size_t o = (size_t)b * DIM + d;
  Xself[o] = (_Float16)x;
  Xn0[o]   = (_Float16)(s0 * (1.0f / FAN));
  Xn1[o]   = (_Float16)(s1 * (1.0f / FAN));
}

// ---------------------------------------------------------------------------
// WMMA fragment helpers.
// ---------------------------------------------------------------------------
__device__ __forceinline__ v16h load_a_frag(const _Float16* p) {
  // 16-bit A 16x32 layout: lane holds K[kh*8 .. +7] then K[kh*8+16 .. +23]
  h8 lo = *(const h8*)p;
  h8 hi = *(const h8*)(p + 16);
  return __builtin_shufflevector(lo, hi, 0, 1, 2, 3, 4, 5, 6, 7,
                                 8, 9, 10, 11, 12, 13, 14, 15);
}

__device__ __forceinline__ v8f wmma_f16(v16h a, v16h b, v8f c) {
  return __builtin_amdgcn_wmma_f32_16x16x32_f16(false, a, false, b,
                                                (short)0, c, false, false);
}

// One (A-pair, B-pair) matmul phase over K=256: 8 k-steps, 4 WMMAs per step
// (2 M-tiles x 2 N-tiles share fragments -> 2 loads per WMMA).
__device__ __forceinline__ void mm_phase(
    const _Float16* __restrict__ a0, const _Float16* __restrict__ a1,
    const _Float16* __restrict__ b0, const _Float16* __restrict__ b1,
    v8f& c00, v8f& c01, v8f& c10, v8f& c11) {
#pragma unroll
  for (int ks = 0; ks < 8; ++ks) {
    v16h af0 = load_a_frag(a0 + ks * 32);
    v16h af1 = load_a_frag(a1 + ks * 32);
    v16h bf0 = *(const v16h*)(b0 + ks * 512);
    v16h bf1 = *(const v16h*)(b1 + ks * 512);
    c00 = wmma_f16(af0, bf0, c00);
    c01 = wmma_f16(af0, bf1, c01);
    c10 = wmma_f16(af1, bf0, c10);
    c11 = wmma_f16(af1, bf1, c11);
  }
}

// ---------------------------------------------------------------------------
// Fused SAGE layer GEMM: out = act(A0 @ W0 + A1 @ W1 + bias).
// One wave32 per 32x32 output block (2x2 WMMA tiles, 4 f32 accumulators).
// K=256 per matmul -> 64 v_wmma_f32_16x16x32_f16 per wave across both
// matmuls. A from row-major f16 activations, B from pre-swizzled weights.
// ---------------------------------------------------------------------------
template <bool RELU, typename OutT>
__global__ __launch_bounds__(256) void sage_gemm(
    const _Float16* __restrict__ A0, const _Float16* __restrict__ A1,
    const _Float16* __restrict__ B0, const _Float16* __restrict__ B1,
    const float* __restrict__ bias, OutT* __restrict__ out) {
  int lane = threadIdx.x & 31;
  int wave = threadIdx.x >> 5;
  int grp  = blockIdx.x * 8 + wave;   // 2048 groups of 2x2 tiles
  int tng  = grp & 7;                 // N groups: DIM/32 = 8
  int tmg  = grp >> 3;                // M groups: BATCH/32 = 256
  int kh   = lane >> 4;

  int m0 = tmg * 32 + (lane & 15);    // first M tile row for this lane
  const _Float16* a0s = A0 + (size_t)m0 * DIM + kh * 8;        // self act, tile m
  const _Float16* a1s = a0s + 16 * DIM;                        // self act, tile m+1
  const _Float16* a0n = A1 + (size_t)m0 * DIM + kh * 8;        // neigh act
  const _Float16* a1n = a0n + 16 * DIM;

  size_t boff = ((size_t)(tng * 2) * 8 * 32 + lane) * 16;      // tn0 fragment base
  const _Float16* b0s = B0 + boff;                             // self W, tile n
  const _Float16* b1s = b0s + 8 * 32 * 16;                     // self W, tile n+1
  const _Float16* b0n = B1 + boff;                             // neigh W
  const _Float16* b1n = b0n + 8 * 32 * 16;

  v8f c00 = {}, c01 = {}, c10 = {}, c11 = {};
  mm_phase(a0s, a1s, b0s, b1s, c00, c01, c10, c11);   // + Xself @ Wself
  mm_phase(a0n, a1n, b0n, b1n, c00, c01, c10, c11);   // + Xneigh @ Wneigh

  // C/D layout: lane -> N = lane&15; VGPR r -> M = (lane>>4)*8 + r.
  int ncol0 = tng * 32 + (lane & 15);
  int ncol1 = ncol0 + 16;
  int mbase = tmg * 32 + kh * 8;
  float bv0 = bias[ncol0];
  float bv1 = bias[ncol1];
#pragma unroll
  for (int r = 0; r < 8; ++r) {
    float v00 = c00[r] + bv0;
    float v01 = c01[r] + bv1;
    float v10 = c10[r] + bv0;
    float v11 = c11[r] + bv1;
    if (RELU) {
      v00 = v00 > 0.f ? v00 : 0.f;
      v01 = v01 > 0.f ? v01 : 0.f;
      v10 = v10 > 0.f ? v10 : 0.f;
      v11 = v11 > 0.f ? v11 : 0.f;
    }
    size_t row0 = (size_t)(mbase + r) * DIM;
    size_t row1 = (size_t)(mbase + 16 + r) * DIM;
    out[row0 + ncol0] = (OutT)v00;
    out[row0 + ncol1] = (OutT)v01;
    out[row1 + ncol0] = (OutT)v10;
    out[row1 + ncol1] = (OutT)v11;
  }
}

extern "C" void kernel_launch(void* const* d_in, const int* in_sizes, int n_in,
                              void* d_out, int out_size, void* d_ws, size_t ws_size,
                              hipStream_t stream) {
  const int*   nodes    = (const int*)d_in[0];
  const int*   neighbor = (const int*)d_in[1];
  const float* ftab     = (const float*)d_in[2];
  const float* store0   = (const float*)d_in[3];
  const float* w_self0  = (const float*)d_in[4];
  const float* w_neigh0 = (const float*)d_in[5];
  const float* b0       = (const float*)d_in[6];
  const float* w_self1  = (const float*)d_in[7];
  const float* w_neigh1 = (const float*)d_in[8];
  const float* b1       = (const float*)d_in[9];

  char* ws = (char*)d_ws;
  size_t off = 0;
  auto alloc = [&](size_t bytes) -> void* {
    void* p = ws + off;
    off += (bytes + 255) & ~(size_t)255;
    return p;
  };
  _Float16* Xself = (_Float16*)alloc((size_t)BATCH * DIM * 2);
  _Float16* Xn0   = (_Float16*)alloc((size_t)BATCH * DIM * 2);
  _Float16* Xn1   = (_Float16*)alloc((size_t)BATCH * DIM * 2);
  _Float16* H     = (_Float16*)alloc((size_t)BATCH * DIM * 2);
  _Float16* Wp0   = (_Float16*)alloc((size_t)DIM * DIM * 2);
  _Float16* Wp1   = (_Float16*)alloc((size_t)DIM * DIM * 2);
  _Float16* Wp2   = (_Float16*)alloc((size_t)DIM * DIM * 2);
  _Float16* Wp3   = (_Float16*)alloc((size_t)DIM * DIM * 2);

  // 1) swizzle+cast weights into WMMA B-fragment layout (4 * 64K halves)
  prep_weights<<<(4 * DIM * DIM) / 256, 256, 0, stream>>>(
      w_self0, w_neigh0, w_self1, w_neigh1, Wp0, Wp1, Wp2, Wp3);

  // 2) gathers + fan-in means (HBM-bound: ~185 MB streamed)
  gather_mean<<<BATCH, DIM, 0, stream>>>(nodes, neighbor, ftab, store0,
                                         Xself, Xn0, Xn1);

  // 3) layer 0: H = relu(Xself@Ws0 + Xn0@Wn0 + b0)  -> f16
  //    2048 wave-groups (2x2 tiles each) / 8 waves per block = 256 blocks
  sage_gemm<true, _Float16><<<256, 256, 0, stream>>>(Xself, Xn0, Wp0, Wp1, b0, H);

  // 4) layer 1: out = H@Ws1 + Xn1@Wn1 + b1          -> f32
  sage_gemm<false, float><<<256, 256, 0, stream>>>(H, Xn1, Wp2, Wp3, b1,
                                                   (float*)d_out);
}
